// SimpleAttention_31997506355478
// MI455X (gfx1250) — matrix operand order repaired and checked by reading, not compiled, not run on hardware
//
#include <hip/hip_runtime.h>
#include <math.h>

typedef __attribute__((ext_vector_type(16))) __bf16 v16bf;
typedef __attribute__((ext_vector_type(8)))  float  v8f;

#define B_   8
#define L_   2048
#define D_   128
#define QT   16            // queries per wave
#define NW   4             // waves per workgroup
#define KB   32            // keys per block iteration
#define NCHUNK (L_ / KB)   // 64
#define VSTR 40            // halves per V^T[d][key] row   (32 keys + 8 pad)  = 80B,  16B aligned
#define KSTR 136           // halves per V [key][d] row    (128 d  + 8 pad)   = 272B, 16B aligned

union BF8 { unsigned u[8]; v16bf v; };

// f32 pair -> packed bf16 (truncating) : single v_perm_b32
static __device__ __forceinline__ unsigned pkbf(float lo, float hi) {
  return __builtin_amdgcn_perm(__float_as_uint(hi), __float_as_uint(lo), 0x07060302u);
}

// cooperative staging: each of 128 threads owns a key-pair x 16-d slice of the
// 32x128 V chunk.  load: 8x global b128.  store: bf16-pack into both layouts.
static __device__ __forceinline__ void stage_load(const float* __restrict__ Vb, int k0, int tid,
                                                  float* va, float* vc) {
  const int key2  = (tid & 15) * 2;
  const int dbase = (tid >> 4) * 16;
  const float* ra = Vb + (size_t)(k0 + key2) * D_ + dbase;
  const float* rb = ra + D_;
  #pragma unroll
  for (int i = 0; i < 16; i += 4) {
    const float4 a = *(const float4*)(ra + i);
    const float4 c = *(const float4*)(rb + i);
    va[i] = a.x; va[i+1] = a.y; va[i+2] = a.z; va[i+3] = a.w;
    vc[i] = c.x; vc[i+1] = c.y; vc[i+2] = c.z; vc[i+3] = c.w;
  }
}

static __device__ __forceinline__ void stage_store(unsigned short* vt, unsigned short* vk, int tid,
                                                   const float* va, const float* vc) {
  const int key2  = (tid & 15) * 2;
  const int dbase = (tid >> 4) * 16;
  unsigned* dvt = (unsigned*)vt;
  unsigned* dvk = (unsigned*)vk;
  #pragma unroll
  for (int i = 0; i < 16; ++i)            // [d][key]: pack across the key pair
    dvt[(dbase + i) * (VSTR / 2) + (key2 >> 1)] = pkbf(va[i], vc[i]);
  #pragma unroll
  for (int i = 0; i < 16; i += 2) {       // [key][d]: pack along d
    dvk[key2       * (KSTR / 2) + ((dbase + i) >> 1)] = pkbf(va[i], va[i+1]);
    dvk[(key2 + 1) * (KSTR / 2) + ((dbase + i) >> 1)] = pkbf(vc[i], vc[i+1]);
  }
}

__launch_bounds__(NW * 32, 1)
__global__ void fa_bf16_wmma_kernel(const float* __restrict__ Qg,
                                    const float* __restrict__ Vg,
                                    float* __restrict__ Og)
{
  __shared__ unsigned short ldsVT[2][D_ * VSTR];   // ping-pong V^T bf16 [d][key]  (PV A)
  __shared__ unsigned short ldsVK[2][KB * KSTR];   // ping-pong V   bf16 [key][d]  (QK A)

  const int tid  = threadIdx.x;
  const int lane = tid & 31;
  const int wave = tid >> 5;
  const int l16  = lane & 15;
  const bool hi  = lane >= 16;

  const int nqb   = L_ / (QT * NW);             // query-blocks per batch = 32
  const int b     = blockIdx.x / nqb;
  const int qblk  = blockIdx.x % nqb;
  const int qbase = qblk * (QT * NW) + wave * QT;

  const float* Qb = Qg + ((size_t)b * L_ + qbase) * D_;
  const float* Vb = Vg + (size_t)b * L_ * D_;

  // ---- preload Q tile as 4 B-operands (B[K=d][N=query], K chunks of 32) ----
  BF8 qop[4];
  {
    const float* qrow = Qb + (size_t)l16 * D_ + (hi ? 8 : 0);
    #pragma unroll
    for (int kk = 0; kk < 4; ++kk) {
      const float4 f0 = *(const float4*)(qrow + kk * 32 + 0);
      const float4 f1 = *(const float4*)(qrow + kk * 32 + 4);
      const float4 f2 = *(const float4*)(qrow + kk * 32 + 16);
      const float4 f3 = *(const float4*)(qrow + kk * 32 + 20);
      qop[kk].u[0] = pkbf(f0.x, f0.y); qop[kk].u[1] = pkbf(f0.z, f0.w);
      qop[kk].u[2] = pkbf(f1.x, f1.y); qop[kk].u[3] = pkbf(f1.z, f1.w);
      qop[kk].u[4] = pkbf(f2.x, f2.y); qop[kk].u[5] = pkbf(f2.z, f2.w);
      qop[kk].u[6] = pkbf(f3.x, f3.y); qop[kk].u[7] = pkbf(f3.z, f3.w);
    }
  }

  // O^T accumulators: 8 d-tiles of 16, C-layout (lane = query column)
  v8f acc[8];
  #pragma unroll
  for (int d = 0; d < 8; ++d)
    #pragma unroll
    for (int r = 0; r < 8; ++r) acc[d][r] = 0.f;

  float m_run = -3.0e38f;
  float l_run = 0.f;

  // ---- pipeline prologue: stage chunk 0 into buffer 0 ----
  {
    float va[16], vc[16];
    stage_load(Vb, 0, tid, va, vc);
    stage_store(ldsVT[0], ldsVK[0], tid, va, vc);
  }
  __syncthreads();

  for (int i = 0; i < NCHUNK; ++i) {
    const int k0  = i * KB;
    const int cur = i & 1;
    const bool have_next = (i + 1 < NCHUNK);

    // issue next chunk's global loads first: latency hidden behind 16 WMMAs
    float va[16], vc[16];
    if (have_next) {
      stage_load(Vb, k0 + KB, tid, va, vc);
      if (i + 2 < NCHUNK)   // global_prefetch_b8 one further ahead
        __builtin_prefetch(Vb + (size_t)(k0 + 2 * KB + (tid & 15) * 2) * D_ + (tid >> 4) * 16, 0, 1);
    }

    // ---- S^T = V_chunk x Q^T : two 16-key tiles, 4 K-chunks of 32 along d ----
    v8f st0, st1;
    #pragma unroll
    for (int r = 0; r < 8; ++r) { st0[r] = 0.f; st1[r] = 0.f; }

    #pragma unroll
    for (int kk = 0; kk < 4; ++kk) {
      const int off = kk * 32 + (hi ? 8 : 0);   // d offset within V row
      {  // keys k0 .. k0+15   (A[m=key][K=d] from LDS, no conversion)
        const unsigned short* p = &ldsVK[cur][(size_t)l16 * KSTR + off];
        BF8 a;
        const uint4 lo4 = *(const uint4*)(p);        // d off .. off+7
        const uint4 hi4 = *(const uint4*)(p + 16);   // d off+16 .. off+23
        a.u[0] = lo4.x; a.u[1] = lo4.y; a.u[2] = lo4.z; a.u[3] = lo4.w;
        a.u[4] = hi4.x; a.u[5] = hi4.y; a.u[6] = hi4.z; a.u[7] = hi4.w;
        st0 = __builtin_amdgcn_wmma_f32_16x16x32_bf16(false, a.v, false, qop[kk].v,
                                                      (short)0, st0, false, false);
      }
      {  // keys k0+16 .. k0+31
        const unsigned short* p = &ldsVK[cur][(size_t)(16 + l16) * KSTR + off];
        BF8 a;
        const uint4 lo4 = *(const uint4*)(p);
        const uint4 hi4 = *(const uint4*)(p + 16);
        a.u[0] = lo4.x; a.u[1] = lo4.y; a.u[2] = lo4.z; a.u[3] = lo4.w;
        a.u[4] = hi4.x; a.u[5] = hi4.y; a.u[6] = hi4.z; a.u[7] = hi4.w;
        st1 = __builtin_amdgcn_wmma_f32_16x16x32_bf16(false, a.v, false, qop[kk].v,
                                                      (short)0, st1, false, false);
      }
    }

    // ---- online softmax for query q = lane&15 (32 scores split lane q / q+16) ----
    float mloc = -3.0e38f;
    #pragma unroll
    for (int r = 0; r < 8; ++r) { mloc = fmaxf(mloc, st0[r]); mloc = fmaxf(mloc, st1[r]); }
    mloc = fmaxf(mloc, __shfl_xor(mloc, 16, 32));
    const float m_new = fmaxf(m_run, mloc);

    if (__any(mloc > m_run)) {       // rare after warm-up: skip rescale when no new max
      const float scale = __expf(m_run - m_new);
      l_run *= scale;
      #pragma unroll
      for (int d = 0; d < 8; ++d)
        #pragma unroll
        for (int r = 0; r < 8; ++r) acc[d][r] *= scale;
    }
    m_run = m_new;

    float p0[8], p1[8], psum = 0.f;
    #pragma unroll
    for (int r = 0; r < 8; ++r) {
      p0[r] = __expf(st0[r] - m_new); psum += p0[r];
      p1[r] = __expf(st1[r] - m_new); psum += p1[r];
    }
    psum += __shfl_xor(psum, 16, 32);
    l_run += psum;

    // ---- build P as B-operand: B[K=key][N=query] ----
    BF8 pop;
    float sel[16];
    #pragma unroll
    for (int r = 0; r < 8; ++r) {
      const float o0 = __shfl_xor(p0[r], 16, 32);   // partner's st0 (keys 8..15 on low)
      const float o1 = __shfl_xor(p1[r], 16, 32);   // partner's st1 (keys 16..23 on high)
      sel[r]     = hi ? o1 : p0[r];
      sel[8 + r] = hi ? p1[r] : o0;
    }
    #pragma unroll
    for (int j = 0; j < 8; ++j) pop.u[j] = pkbf(sel[2 * j], sel[2 * j + 1]);

    // ---- O^T += V^T x P : 8 d-tiles; A from LDS (b128, conflict-free stride) ----
    const int keyoff = hi ? 16 : 0;
    #pragma unroll
    for (int dt = 0; dt < 8; ++dt) {
      const unsigned short* row = &ldsVT[cur][(dt * 16 + l16) * VSTR + keyoff];
      BF8 a;
      const uint4 lo4 = *(const uint4*)(row);        // keys keyoff .. keyoff+7
      const uint4 hi4 = *(const uint4*)(row + 8);    // keys keyoff+8 .. keyoff+15
      a.u[0] = lo4.x; a.u[1] = lo4.y; a.u[2] = lo4.z; a.u[3] = lo4.w;
      a.u[4] = hi4.x; a.u[5] = hi4.y; a.u[6] = hi4.z; a.u[7] = hi4.w;
      acc[dt] = __builtin_amdgcn_wmma_f32_16x16x32_bf16(false, a.v, false, pop.v,
                                                        (short)0, acc[dt], false, false);
    }

    // ---- stage next chunk into the other buffer, then single barrier ----
    if (have_next)
      stage_store(ldsVT[cur ^ 1], ldsVK[cur ^ 1], tid, va, vc);
    __syncthreads();
  }

  // ---- epilogue: O[q][d] = acc^T / l_run ----
  const float inv = 1.f / l_run;
  float* orow = Og + ((size_t)b * L_ + qbase + l16) * D_;
  #pragma unroll
  for (int dt = 0; dt < 8; ++dt) {
    const int d0 = dt * 16 + (hi ? 8 : 0);
    float4 f0, f1;
    f0.x = acc[dt][0] * inv; f0.y = acc[dt][1] * inv;
    f0.z = acc[dt][2] * inv; f0.w = acc[dt][3] * inv;
    f1.x = acc[dt][4] * inv; f1.y = acc[dt][5] * inv;
    f1.z = acc[dt][6] * inv; f1.w = acc[dt][7] * inv;
    *(float4*)(orow + d0)     = f0;
    *(float4*)(orow + d0 + 4) = f1;
  }
}

extern "C" void kernel_launch(void* const* d_in, const int* in_sizes, int n_in,
                              void* d_out, int out_size, void* d_ws, size_t ws_size,
                              hipStream_t stream) {
  (void)in_sizes; (void)n_in; (void)d_ws; (void)ws_size; (void)out_size;
  const float* Q = (const float*)d_in[0];
  const float* V = (const float*)d_in[1];
  float* O = (float*)d_out;
  dim3 grid(B_ * (L_ / (QT * NW)));   // 8 * 32 = 256 workgroups
  dim3 block(NW * 32);                // 128 threads = 4 waves
  hipLaunchKernelGGL(fa_bf16_wmma_kernel, grid, block, 0, stream, Q, V, O);
}
